// AttnDecoderRNN_27736898797807
// MI455X (gfx1250) — matrix operand order repaired and checked by reading, not compiled, not run on hardware
//
#include <hip/hip_runtime.h>
#include <math.h>

// Problem dims
#define H  1024
#define H2 2048
#define V  50257
#define L  512

typedef __attribute__((ext_vector_type(16))) _Float16 v16h;
typedef __attribute__((ext_vector_type(8)))  float    v8f;

// wave32 sum reduction (lane 0 holds result)
__device__ __forceinline__ float wave_sum(float v) {
#pragma unroll
  for (int off = 16; off > 0; off >>= 1)
    v += __shfl_down(v, off, 32);
  return v;
}

// ---------------------------------------------------------------------------
// K1: attn_in = [ emb[token], hidden ]   (2048 floats)
__global__ void k_embed(const int* __restrict__ tok, const float* __restrict__ emb,
                        const float* __restrict__ hidden, float* __restrict__ attn_in) {
  int t = blockIdx.x * blockDim.x + threadIdx.x;
  if (t < H)       attn_in[t] = emb[(size_t)tok[0] * H + t];
  else if (t < H2) attn_in[t] = hidden[t - H];
}

// ---------------------------------------------------------------------------
// K2: scores[l] = dot(attn_W[l,:], attn_in) + attn_b[l]   (one wave per row)
__global__ void k_attn_scores(const float* __restrict__ attn_in, const float* __restrict__ attn_W,
                              const float* __restrict__ attn_b, float* __restrict__ scores) {
  int wid  = (blockIdx.x * blockDim.x + threadIdx.x) >> 5;   // row, 0..511
  int lane = threadIdx.x & 31;
  const float4* Wr = (const float4*)(attn_W + (size_t)wid * H2);
  const float4* Xr = (const float4*)attn_in;
  float acc = 0.f;
#pragma unroll
  for (int i = 0; i < 16; ++i) {
    float4 w4 = Wr[i * 32 + lane];
    float4 x4 = Xr[i * 32 + lane];
    acc += w4.x * x4.x + w4.y * x4.y + w4.z * x4.z + w4.w * x4.w;
  }
  acc = wave_sum(acc);
  if (lane == 0) scores[wid] = acc + attn_b[wid];
}

// ---------------------------------------------------------------------------
// K3: softmax over L=512 (single block), write attn_weights to d_out section
__global__ void k_softmax(const float* __restrict__ scores, float* __restrict__ attn_w_out) {
  __shared__ float red[L];
  int t = threadIdx.x;                 // 512 threads
  float v = scores[t];
  red[t] = v;
  __syncthreads();
  for (int s = 256; s > 0; s >>= 1) { if (t < s) red[t] = fmaxf(red[t], red[t + s]); __syncthreads(); }
  float m = red[0];
  __syncthreads();
  float e = expf(v - m);
  red[t] = e;
  __syncthreads();
  for (int s = 256; s > 0; s >>= 1) { if (t < s) red[t] += red[t + s]; __syncthreads(); }
  attn_w_out[t] = e / red[0];
}

// ---------------------------------------------------------------------------
// K4 (WMMA): attn_applied partials.  D = A x B + C with
//   A row0 = attn_w[k0..k0+31] (f16), rows 1..15 = 0
//   B[k][n] = enc[(k0+k)*H + c0+n]  -- B rows are contiguous in memory, so the
//   per-lane layout (lane = column n, halves of the wave = K halves) gives
//   64B-coalesced segments.  256 waves: 64 column tiles x 4 row segments.
__global__ void k_attn_applied_wmma(const float* __restrict__ attn_w, const float* __restrict__ enc,
                                    float* __restrict__ partial) {
  int wid    = (blockIdx.x * blockDim.x + threadIdx.x) >> 5;  // 0..255
  int lane   = threadIdx.x & 31;
  int tile   = wid >> 2;            // 0..63 -> columns c0..c0+15
  int seg    = wid & 3;             // 0..3  -> rows  seg*128 .. +127
  int c0     = tile * 16;
  int col    = c0 + (lane & 15);
  int laneHi = lane >> 4;           // 0: lanes 0-15, 1: lanes 16-31

  v8f acc;
#pragma unroll
  for (int i = 0; i < 8; ++i) acc[i] = 0.f;

#pragma unroll
  for (int step = 0; step < 4; ++step) {
    int k0 = seg * 128 + step * 32;

    // A operand (16x32 f16): lane m holds row m; only row 0 is non-zero.
    // Lane 0: elems 0-15 = K0..15; lane 16: elems 0-15 = K16..31.
    v16h a;
#pragma unroll
    for (int i = 0; i < 16; ++i) a[i] = (_Float16)0.f;
    if ((lane & 15) == 0) {
      int base = k0 + laneHi * 16;
#pragma unroll
      for (int i = 0; i < 8; ++i) {
        a[i]     = (_Float16)attn_w[base + i];
        a[8 + i] = (_Float16)attn_w[base + 8 + i];
      }
    }

    // B operand (32x16 f16): lanes 0-15 hold K=0..15 (elem e = row e), lanes
    // 16-31 hold K=16..31; lane index inside half = column.
    v16h b;
    int rbase = k0 + laneHi * 16;
#pragma unroll
    for (int e = 0; e < 16; ++e)
      b[e] = (_Float16)enc[(size_t)(rbase + e) * H + col];

    acc = __builtin_amdgcn_wmma_f32_16x16x32_f16(
        /*neg_a=*/false, a, /*neg_b=*/false, b,
        /*c_mod=*/(short)0, acc, /*reuse_a=*/false, /*reuse_b=*/false);
  }
  // D row 0 lives in acc[0] of lanes 0-15 (VGPR0, M=0, N=lane)
  if (lane < 16) partial[seg * H + c0 + lane] = acc[0];
}

// ---------------------------------------------------------------------------
// K5: x2 = [ embedded, sum of 4 attn_applied partials ]
__global__ void k_build_x2(const float* __restrict__ attn_in, const float* __restrict__ partial,
                           float* __restrict__ x2) {
  int t = blockIdx.x * blockDim.x + threadIdx.x;  // 0..1023
  x2[t]     = attn_in[t];
  x2[H + t] = partial[t] + partial[H + t] + partial[2 * H + t] + partial[3 * H + t];
}

// ---------------------------------------------------------------------------
// K6: x = relu(comb_W @ x2 + comb_b)   (one wave per output row)
__global__ void k_combine(const float* __restrict__ x2, const float* __restrict__ comb_W,
                          const float* __restrict__ comb_b, float* __restrict__ x) {
  int wid  = (blockIdx.x * blockDim.x + threadIdx.x) >> 5;   // 0..1023
  int lane = threadIdx.x & 31;
  const float4* Wr = (const float4*)(comb_W + (size_t)wid * H2);
  const float4* Xr = (const float4*)x2;
  float acc = 0.f;
#pragma unroll
  for (int i = 0; i < 16; ++i) {
    float4 w4 = Wr[i * 32 + lane];
    float4 x4 = Xr[i * 32 + lane];
    acc += w4.x * x4.x + w4.y * x4.y + w4.z * x4.z + w4.w * x4.w;
  }
  acc = wave_sum(acc);
  if (lane == 0) x[wid] = fmaxf(acc + comb_b[wid], 0.f);
}

// ---------------------------------------------------------------------------
// K7: GRU step; wave j computes all six 1024-dots for output element j.
__global__ void k_gru(const float* __restrict__ x, const float* __restrict__ hidden,
                      const float* __restrict__ w_ih, const float* __restrict__ w_hh,
                      const float* __restrict__ b_ih, const float* __restrict__ b_hh,
                      float* __restrict__ h_new) {
  int j    = (blockIdx.x * blockDim.x + threadIdx.x) >> 5;   // 0..1023
  int lane = threadIdx.x & 31;
  const float4* xr  = (const float4*)x;
  const float4* hr  = (const float4*)hidden;
  const float4* wir = (const float4*)(w_ih + (size_t)j * H);
  const float4* wiz = (const float4*)(w_ih + (size_t)(H + j) * H);
  const float4* win = (const float4*)(w_ih + (size_t)(2 * H + j) * H);
  const float4* whr = (const float4*)(w_hh + (size_t)j * H);
  const float4* whz = (const float4*)(w_hh + (size_t)(H + j) * H);
  const float4* whn = (const float4*)(w_hh + (size_t)(2 * H + j) * H);
  float air = 0.f, aiz = 0.f, ain = 0.f, ahr = 0.f, ahz = 0.f, ahn = 0.f;
#pragma unroll
  for (int i = 0; i < 8; ++i) {
    int idx = i * 32 + lane;
    float4 xv = xr[idx], hv = hr[idx], w;
    w = wir[idx]; air += w.x * xv.x + w.y * xv.y + w.z * xv.z + w.w * xv.w;
    w = wiz[idx]; aiz += w.x * xv.x + w.y * xv.y + w.z * xv.z + w.w * xv.w;
    w = win[idx]; ain += w.x * xv.x + w.y * xv.y + w.z * xv.z + w.w * xv.w;
    w = whr[idx]; ahr += w.x * hv.x + w.y * hv.y + w.z * hv.z + w.w * hv.w;
    w = whz[idx]; ahz += w.x * hv.x + w.y * hv.y + w.z * hv.z + w.w * hv.w;
    w = whn[idx]; ahn += w.x * hv.x + w.y * hv.y + w.z * hv.z + w.w * hv.w;
  }
  air = wave_sum(air); aiz = wave_sum(aiz); ain = wave_sum(ain);
  ahr = wave_sum(ahr); ahz = wave_sum(ahz); ahn = wave_sum(ahn);
  if (lane == 0) {
    float i_r = air + b_ih[j],         h_r = ahr + b_hh[j];
    float i_z = aiz + b_ih[H + j],     h_z = ahz + b_hh[H + j];
    float i_n = ain + b_ih[2 * H + j], h_n = ahn + b_hh[2 * H + j];
    float r = 1.f / (1.f + expf(-(i_r + h_r)));
    float z = 1.f / (1.f + expf(-(i_z + h_z)));
    float n = tanhf(i_n + r * h_n);
    h_new[j] = (1.f - z) * n + z * hidden[j];
  }
}

// ---------------------------------------------------------------------------
// K8: logits[v] = dot(out_W[v,:], h_new) + out_b[v].  206 MB stream: one wave
// per vocab row, b128 coalesced loads; this is the HBM-roofline stage.
__global__ void k_logits(const float* __restrict__ h_new, const float* __restrict__ out_W,
                         const float* __restrict__ out_b, float* __restrict__ logits) {
  int v = (blockIdx.x * blockDim.x + threadIdx.x) >> 5;
  if (v >= V) return;
  int lane = threadIdx.x & 31;
  const float4* Wr = (const float4*)(out_W + (size_t)v * H);
  const float4* hv = (const float4*)h_new;
  if (lane == 0 && v + 64 < V)                   // global_prefetch_b8, keep HBM pipe full
    __builtin_prefetch(out_W + (size_t)(v + 64) * H, 0, 1);
  float acc = 0.f;
#pragma unroll
  for (int i = 0; i < 8; ++i) {
    float4 w4 = Wr[i * 32 + lane];
    float4 h4 = hv[i * 32 + lane];
    acc += w4.x * h4.x + w4.y * h4.y + w4.z * h4.z + w4.w * h4.w;
  }
  acc = wave_sum(acc);
  if (lane == 0) logits[v] = acc + out_b[v];
}

// ---------------------------------------------------------------------------
// K9/K10/K11: deterministic two-phase log-softmax over V
__global__ void k_lse_part(const float* __restrict__ logits, float* __restrict__ red) {
  __shared__ float sm[256];
  int b = blockIdx.x, t = threadIdx.x;
  int start = b * 786;
  int end = start + 786; if (end > V) end = V;
  float m = -1e30f;
  for (int i = start + t; i < end; i += 256) m = fmaxf(m, logits[i]);
  sm[t] = m; __syncthreads();
  for (int s = 128; s > 0; s >>= 1) { if (t < s) sm[t] = fmaxf(sm[t], sm[t + s]); __syncthreads(); }
  float M = sm[0]; __syncthreads();
  float sum = 0.f;
  for (int i = start + t; i < end; i += 256) sum += expf(logits[i] - M);
  sm[t] = sum; __syncthreads();
  for (int s = 128; s > 0; s >>= 1) { if (t < s) sm[t] += sm[t + s]; __syncthreads(); }
  if (t == 0) { red[2 * b] = M; red[2 * b + 1] = sm[0]; }
}

__global__ void k_lse_final(const float* __restrict__ red, float* __restrict__ mf) {
  float M = -1e30f;
  for (int i = 0; i < 64; ++i) M = fmaxf(M, red[2 * i]);
  float S = 0.f;
  for (int i = 0; i < 64; ++i) S += red[2 * i + 1] * expf(red[2 * i] - M);
  mf[0] = M; mf[1] = logf(S);
}

__global__ void k_logsoftmax(const float* __restrict__ logits, const float* __restrict__ mf,
                             float* __restrict__ out) {
  int i = blockIdx.x * blockDim.x + threadIdx.x;
  if (i < V) out[i] = logits[i] - mf[0] - mf[1];
}

// ---------------------------------------------------------------------------
extern "C" void kernel_launch(void* const* d_in, const int* in_sizes, int n_in,
                              void* d_out_v, int out_size, void* d_ws, size_t ws_size,
                              hipStream_t stream) {
  const int*   tok    = (const int*)  d_in[0];
  const float* hidden = (const float*)d_in[1];
  const float* enc    = (const float*)d_in[2];
  const float* emb    = (const float*)d_in[3];
  const float* attn_W = (const float*)d_in[4];
  const float* attn_b = (const float*)d_in[5];
  const float* comb_W = (const float*)d_in[6];
  const float* comb_b = (const float*)d_in[7];
  const float* gwih   = (const float*)d_in[8];
  const float* gwhh   = (const float*)d_in[9];
  const float* gbih   = (const float*)d_in[10];
  const float* gbhh   = (const float*)d_in[11];
  const float* out_W  = (const float*)d_in[12];
  const float* out_b  = (const float*)d_in[13];

  float* out       = (float*)d_out_v;
  float* out_logp  = out;            // [V]   log-softmax output
  float* out_hnew  = out + V;        // [H]   new hidden
  float* out_attnw = out + V + H;    // [L]   attention weights

  float* ws         = (float*)d_ws;
  float* ws_attn_in = ws;            // 2048
  float* ws_scores  = ws + 2048;     // 512
  float* ws_partial = ws + 2560;     // 4*1024
  float* ws_x2      = ws + 6656;     // 2048
  float* ws_x       = ws + 8704;     // 1024
  float* ws_logits  = ws + 9728;     // 50257 (ends at 59985)
  float* ws_red     = ws + 60000;    // 128
  float* ws_mf      = ws + 60160;    // 2

  k_embed            <<<8,             256, 0, stream>>>(tok, emb, hidden, ws_attn_in);
  k_attn_scores      <<<64,            256, 0, stream>>>(ws_attn_in, attn_W, attn_b, ws_scores);
  k_softmax          <<<1,             512, 0, stream>>>(ws_scores, out_attnw);
  k_attn_applied_wmma<<<32,            256, 0, stream>>>(out_attnw, enc, ws_partial);
  k_build_x2         <<<4,             256, 0, stream>>>(ws_attn_in, ws_partial, ws_x2);
  k_combine          <<<128,           256, 0, stream>>>(ws_x2, comb_W, comb_b, ws_x);
  k_gru              <<<128,           256, 0, stream>>>(ws_x, hidden, gwih, gwhh, gbih, gbhh, out_hnew);
  k_logits           <<<(V + 7) / 8,   256, 0, stream>>>(out_hnew, out_W, out_b, ws_logits);
  k_lse_part         <<<64,            256, 0, stream>>>(ws_logits, ws_red);
  k_lse_final        <<<1,               1, 0, stream>>>(ws_red, ws_mf);
  k_logsoftmax       <<<(V + 255)/256, 256, 0, stream>>>(ws_logits, ws_mf, out_logp);
}
